// MixtureOfExperts_5514738008955
// MI455X (gfx1250) — matrix operand order repaired and checked by reading, not compiled, not run on hardware
//
#include <hip/hip_runtime.h>
#include <hip/hip_bf16.h>
#include <stdint.h>

// ---------------- problem constants (B=4,S=2048 -> T=8192) ----------------
#define T_TOK 8192
#define DDIM  1024
#define FDIM  4096
#define NEXP  8
#define BM    64      // tokens per FFN block
#define NTHR  512     // 16 waves
#define BF    128     // F-chunk width
#define LDX   1032    // lds_x row stride (bf16 elems): 2064B -> 16B aligned, 4-bank row shift
#define LDH   136     // lds_h row stride (bf16 elems): 272B  -> 16B aligned, 4-bank row shift

typedef __attribute__((ext_vector_type(16))) __bf16       v16bf;
typedef __attribute__((ext_vector_type(8)))  float        v8f;
typedef __attribute__((ext_vector_type(8)))  unsigned int v8u;
typedef __attribute__((ext_vector_type(4)))  unsigned int v4u;

__device__ __forceinline__ unsigned int f2bf(float f) {
  unsigned int u = __builtin_bit_cast(unsigned int, f);
  u += 0x7FFFu + ((u >> 16) & 1u);          // round-to-nearest-even
  return u >> 16;
}

__device__ __forceinline__ v8f wmma_bf16(v16bf a, v16bf b, v8f c) {
  return __builtin_amdgcn_wmma_f32_16x16x32_bf16(false, a, false, b, (short)0, c,
                                                 false, false);
}

// A 16x32 k order used here (contiguous, same as B): lane L (m=L&15): K = {0..15} (L<16) / {16..31}
// -> two 16B ds_load_b128 per fragment.
__device__ __forceinline__ v16bf load_a_lds(const unsigned short* p, int ld,
                                            int row0, int k0, int lane) {
  const int m   = row0 + (lane & 15);
  const int klo = (lane & 16) ? 16 : 0;
  const unsigned short* rp = p + m * ld + k0 + klo;
  v4u lo = *(const v4u*)rp;          // K = klo .. klo+7
  v4u hi = *(const v4u*)(rp + 8);    // K = klo+8 .. klo+15
  v8u u  = __builtin_shufflevector(lo, hi, 0, 1, 2, 3, 4, 5, 6, 7);
  return __builtin_bit_cast(v16bf, u);
}

// B 32x16 k order used here (same as A): lane = n + 16*(K>=16); per lane K = kb+0..15.
// With pre-transposed bf16 weights those 16 K values are CONTIGUOUS: one 32B load.
__device__ __forceinline__ v16bf load_b_bf16(const unsigned short* bp) {
  return __builtin_bit_cast(v16bf, *(const v8u*)bp);
}

// ---------------- kernel 0: zero output + expert counters ----------------
__global__ void moe_init(float* __restrict__ out, int* __restrict__ cnt, int n4) {
  int idx = blockIdx.x * blockDim.x + threadIdx.x;
  if (idx < NEXP) cnt[idx] = 0;
  float4* o4 = (float4*)out;
  float4 z = make_float4(0.f, 0.f, 0.f, 0.f);
  for (int k = idx; k < n4; k += gridDim.x * blockDim.x) o4[k] = z;
}

// ---------- kernel 0b: per-expert fp32 -> bf16 transpose of weights ----------
// src[e][R][C] (fp32, row-major) -> dst[e][C][R] (bf16, row-major)
__global__ __launch_bounds__(256) void transpose_bf16(const float* __restrict__ src,
                                                      unsigned short* __restrict__ dst,
                                                      int R, int C) {
  __shared__ unsigned short tile[32][33];
  const size_t off = (size_t)blockIdx.z * R * C;
  const float* s = src + off;
  unsigned short* d = dst + off;
  const int c0 = blockIdx.x * 32, r0 = blockIdx.y * 32;
  const int tx = threadIdx.x & 31, ty = threadIdx.x >> 5;
#pragma unroll
  for (int i = 0; i < 4; ++i) {
    int row = ty + 8 * i;
    tile[row][tx] = (unsigned short)f2bf(s[(size_t)(r0 + row) * C + (c0 + tx)]);
  }
  __syncthreads();
#pragma unroll
  for (int i = 0; i < 4; ++i) {
    int orow = ty + 8 * i;
    d[(size_t)(c0 + orow) * R + (r0 + tx)] = tile[tx][orow];
  }
}

// ---------------- kernel 1: gating (one wave32 per token) ----------------
__global__ __launch_bounds__(256) void moe_gate(const float* __restrict__ x,
                                                const float* __restrict__ gw,
                                                int* __restrict__ cnt,
                                                int* __restrict__ tok_id,
                                                float* __restrict__ tok_w) {
  const int lane = threadIdx.x & 31;
  const int t = blockIdx.x * 8 + (threadIdx.x >> 5);
  float p[NEXP];
#pragma unroll
  for (int e = 0; e < NEXP; ++e) p[e] = 0.f;

  const float* xr = x + (size_t)t * DDIM;
  for (int i = lane; i < DDIM; i += 32) {
    float xv = xr[i];
    const float4* g4 = (const float4*)(gw + (size_t)i * NEXP);
    float4 ga = g4[0], gb = g4[1];
    p[0] += xv * ga.x; p[1] += xv * ga.y; p[2] += xv * ga.z; p[3] += xv * ga.w;
    p[4] += xv * gb.x; p[5] += xv * gb.y; p[6] += xv * gb.z; p[7] += xv * gb.w;
  }
#pragma unroll
  for (int e = 0; e < NEXP; ++e)
    for (int off = 16; off > 0; off >>= 1) p[e] += __shfl_down(p[e], off, 32);

  if (lane == 0) {
    int i0 = 0; float v0 = p[0];
#pragma unroll
    for (int e = 1; e < NEXP; ++e) if (p[e] > v0) { v0 = p[e]; i0 = e; }
    int i1 = -1; float v1 = -3.402823466e38f;
#pragma unroll
    for (int e = 0; e < NEXP; ++e)
      if (e != i0 && p[e] > v1) { v1 = p[e]; i1 = e; }
    float e1 = __expf(v1 - v0);
    float inv = 1.0f / (1.0f + e1);
    float w0 = inv, w1 = e1 * inv;
    int p0 = atomicAdd(&cnt[i0], 1);
    tok_id[i0 * T_TOK + p0] = t;  tok_w[i0 * T_TOK + p0] = w0;
    int p1 = atomicAdd(&cnt[i1], 1);
    tok_id[i1 * T_TOK + p1] = t;  tok_w[i1 * T_TOK + p1] = w1;
  }
}

// ---------------- kernel 2: fused expert FFN (bf16 WMMA) ----------------
// w1t: [E][F][D] bf16 (W1 transposed), w2t: [E][D][F] bf16 (W2 transposed)
__global__ __launch_bounds__(NTHR) void moe_ffn(const float* __restrict__ x,
                                                const unsigned short* __restrict__ w1t,
                                                const unsigned short* __restrict__ w2t,
                                                const int* __restrict__ cnt,
                                                const int* __restrict__ tok_id,
                                                const float* __restrict__ tok_w,
                                                float* __restrict__ out) {
  const int e  = blockIdx.y;
  const int m0 = blockIdx.x * BM;
  const int cnt_e = cnt[e];
  if (m0 >= cnt_e) return;

  __shared__ __align__(16) unsigned short lds_x[BM * LDX];  // 132096 B
  __shared__ __align__(16) unsigned short lds_h[BM * LDH];  //  17408 B
  __shared__ int   lds_tok[BM];
  __shared__ float lds_wgt[BM];

  const int tid   = threadIdx.x;
  const int lane  = tid & 31;
  const int wv    = tid >> 5;            // 16 waves
  const int kb    = (lane & 16) ? 16 : 0;
  const int nlane = lane & 15;

  if (tid < BM) {
    int r = m0 + tid;
    lds_tok[tid] = (r < cnt_e) ? tok_id[e * T_TOK + r] : 0;
    lds_wgt[tid] = (r < cnt_e) ? tok_w[e * T_TOK + r] : 0.0f;
  }
  __syncthreads();

  { // gather 64 token rows -> LDS bf16 (8 threads per row, 128 cols each)
    int r  = tid >> 3;
    int c0 = (tid & 7) * 128;
    bool valid = (m0 + r) < cnt_e;
    const float4* src = (const float4*)(x + (size_t)lds_tok[r] * DDIM + c0);
#pragma unroll 4
    for (int i = 0; i < 32; ++i) {
      float4 v = valid ? src[i] : make_float4(0.f, 0.f, 0.f, 0.f);
      unsigned int q0 = f2bf(v.x) | (f2bf(v.y) << 16);
      unsigned int q1 = f2bf(v.z) | (f2bf(v.w) << 16);
      *(uint2*)(lds_x + r * LDX + c0 + 4 * i) = make_uint2(q0, q1);
    }
  }
  __syncthreads();

  const v8f zero8 = {0.f, 0.f, 0.f, 0.f, 0.f, 0.f, 0.f, 0.f};
  v8f acc[4][4];                        // wave's 64row x 64col fp32 accumulator
#pragma unroll
  for (int a = 0; a < 4; ++a)
#pragma unroll
    for (int b = 0; b < 4; ++b) acc[a][b] = zero8;

  const unsigned short* W1te = w1t + (size_t)e * DDIM * FDIM;
  const unsigned short* W2te = w2t + (size_t)e * FDIM * DDIM;

  // GEMM1 wave mapping: nw = H col-tile (16 cols), mh = row half (0 or 32)
  const int nw = wv & 7;
  const int mh = (wv >> 3) * 32;
  // GEMM2 wave mapping: 64-column D slice
  const int col0 = 64 * wv;

#pragma unroll 1
  for (int fc = 0; fc < FDIM; fc += BF) {
    // ---- GEMM1: H[mh..mh+32, 16 cols of this wave] = X @ W1[:, fc+16nw..]
    v8f h0 = zero8, h1 = zero8;
    const unsigned short* w1row =
        W1te + (size_t)(fc + 16 * nw + nlane) * DDIM + kb;
#pragma unroll 4
    for (int ks = 0; ks < DDIM / 32; ++ks) {
      const int k0 = ks * 32;
      v16bf b  = load_b_bf16(w1row + k0);
      if (ks + 4 < DDIM / 32) __builtin_prefetch(w1row + k0 + 128, 0, 1);
      v16bf a0 = load_a_lds(lds_x, LDX, mh,      k0, lane);
      v16bf a1 = load_a_lds(lds_x, LDX, mh + 16, k0, lane);
      h0 = wmma_bf16(a0, b, h0);
      h1 = wmma_bf16(a1, b, h1);
    }
    { // SiLU + store H strip (C/D layout: VGPR j -> row j (+8 for lanes>=16))
      int n  = 16 * nw + nlane;
      int rb = (lane & 16) ? 8 : 0;
#pragma unroll
      for (int j = 0; j < 8; ++j) {
        float a = h0[j]; a = a / (1.0f + __expf(-a));
        float c = h1[j]; c = c / (1.0f + __expf(-c));
        lds_h[(mh + rb + j) * LDH + n]      = (unsigned short)f2bf(a);
        lds_h[(mh + 16 + rb + j) * LDH + n] = (unsigned short)f2bf(c);
      }
    }
    __syncthreads();

    // ---- GEMM2: acc(64 x 64 cols of this wave) += H(64x128) @ W2[fc.., :]
    const unsigned short* w2row = W2te + (size_t)(col0 + nlane) * FDIM + fc + kb;
#pragma unroll
    for (int ks = 0; ks < BF / 32; ++ks) {
      const int k0 = ks * 32;
      v16bf a0 = load_a_lds(lds_h, LDH, 0,  k0, lane);
      v16bf a1 = load_a_lds(lds_h, LDH, 16, k0, lane);
      v16bf a2 = load_a_lds(lds_h, LDH, 32, k0, lane);
      v16bf a3 = load_a_lds(lds_h, LDH, 48, k0, lane);
#pragma unroll
      for (int nt = 0; nt < 4; ++nt) {
        v16bf b = load_b_bf16(w2row + (size_t)(16 * nt) * FDIM + k0);
        acc[0][nt] = wmma_bf16(a0, b, acc[0][nt]);
        acc[1][nt] = wmma_bf16(a1, b, acc[1][nt]);
        acc[2][nt] = wmma_bf16(a2, b, acc[2][nt]);
        acc[3][nt] = wmma_bf16(a3, b, acc[3][nt]);
      }
    }
    __syncthreads();   // lds_h rewritten next chunk
  }

  // ---- epilogue: weighted atomic scatter-add into out
  const int rb = (lane & 16) ? 8 : 0;
#pragma unroll
  for (int mt = 0; mt < 4; ++mt) {
#pragma unroll
    for (int j = 0; j < 8; ++j) {
      int r = mt * 16 + rb + j;
      if (m0 + r < cnt_e) {
        float wgt = lds_wgt[r];
        size_t ob = (size_t)lds_tok[r] * DDIM + col0 + nlane;
#pragma unroll
        for (int nt = 0; nt < 4; ++nt) {
          (void)__hip_atomic_fetch_add(out + ob + 16 * nt, wgt * acc[mt][nt][j],
                                       __ATOMIC_RELAXED, __HIP_MEMORY_SCOPE_AGENT);
        }
      }
    }
  }
}

// ---------------- launcher ----------------
extern "C" void kernel_launch(void* const* d_in, const int* in_sizes, int n_in,
                              void* d_out, int out_size, void* d_ws, size_t ws_size,
                              hipStream_t stream) {
  const float* x  = (const float*)d_in[0];   // [4,2048,1024]
  const float* gw = (const float*)d_in[1];   // [1024,8]
  const float* w1 = (const float*)d_in[2];   // [8,1024,4096]
  const float* w2 = (const float*)d_in[3];   // [8,4096,1024]
  float* out = (float*)d_out;                // [4,2048,1024]

  // ws layout: cnt | tok_id | tok_w | w1t (bf16) | w2t (bf16)
  char* ws = (char*)d_ws;
  int*   cnt    = (int*)ws;                                    // 8 ints
  int*   tok_id = (int*)(ws + 256);                            // E*T ints
  float* tok_w  = (float*)(ws + 256 + (size_t)NEXP * T_TOK * 4);
  size_t wtoff  = 256 + 2 * (size_t)NEXP * T_TOK * 4;          // 256-aligned
  unsigned short* w1t = (unsigned short*)(ws + wtoff);         // [E][F][D] bf16
  unsigned short* w2t = w1t + (size_t)NEXP * DDIM * FDIM;      // [E][D][F] bf16

  moe_init<<<2048, 256, 0, stream>>>(out, cnt, (T_TOK * DDIM) / 4);
  transpose_bf16<<<dim3(FDIM / 32, DDIM / 32, NEXP), 256, 0, stream>>>(
      w1, w1t, DDIM, FDIM);
  transpose_bf16<<<dim3(DDIM / 32, FDIM / 32, NEXP), 256, 0, stream>>>(
      w2, w2t, FDIM, DDIM);
  moe_gate<<<T_TOK / 8, 256, 0, stream>>>(x, gw, cnt, tok_id, tok_w);
  moe_ffn<<<dim3(T_TOK / BM, NEXP), NTHR, 0, stream>>>(x, w1t, w2t, cnt, tok_id,
                                                       tok_w, out);
}
